// GCNPolicy_20298015441054
// MI455X (gfx1250) — matrix operand order repaired and checked by reading, not compiled, not run on hardware
//
#include <hip/hip_runtime.h>
#include <math.h>

typedef __attribute__((ext_vector_type(16))) _Float16 v16h;
typedef __attribute__((ext_vector_type(8)))  _Float16 v8h;
typedef __attribute__((ext_vector_type(8)))  float    v8f;

#define N_NODES 8
#define OBS     32
#define LSEQ    5
#define TBATCH  8
#define ROWS    (TBATCH * N_NODES)   // 64 node-rows per workgroup
#define SROWS   (ROWS * 3)           // 192 conv1 super-rows (row, t)
#define XSTR    160                  // x tile: [row][t'(5)][obs(32)], 16B-aligned rows
#define Y1STR   72                   // y1 stride (64 ch + pad, multiple of 8)
#define ZSTR    72                   // y2 / Z stride
#define HSTR    136                  // 128 ch + pad (multiple of 8)

// ---- WMMA f16 fragment index helpers (wave32, 16x16x32) ----
// A (16x32): lane holds row (lane&15); halves 0-7 = K [lanehi*8 .. +7],
//            halves 8-15 = K [16+lanehi*8 .. +7]  -> two contiguous 16B chunks.
// B (32x16): lane holds col (lane&15); halves 0-7 = K [lanehi*8 .. +7],
//            halves 8-15 = K [16+lanehi*8 .. +7] (same order as A).
#define KFB(j, h) ((lanehi << 3) + (((j) >> 2) << 3) + ((j) << 1) + (h))

// A-fragment from LDS: p points at K=0 of this 32-wide K-step for this lane's row.
#define LOAD_AFRAG(p) ({                                                       \
    v8h lo_ = *(const v8h*)((p) + (lanehi << 3));                              \
    v8h hi_ = *(const v8h*)((p) + 16 + (lanehi << 3));                         \
    __builtin_shufflevector(lo_, hi_, 0,1,2,3,4,5,6,7,8,9,10,11,12,13,14,15); })

static __device__ __forceinline__ float2 h2f2(unsigned int u) {
    union { unsigned int u; _Float16 h[2]; } x; x.u = u;
    return make_float2((float)x.h[0], (float)x.h[1]);
}
static __device__ __forceinline__ unsigned int f2h2(float a, float b) {
    union { unsigned int u; _Float16 h[2]; } x;
    x.h[0] = (_Float16)a; x.h[1] = (_Float16)b; return x.u;
}

__global__ __launch_bounds__(256)
void gcn_policy_fused(const float* __restrict__ data,
                      const float* __restrict__ w1,  const float* __restrict__ b1,
                      const float* __restrict__ w2,  const float* __restrict__ b2,
                      const float* __restrict__ gw1, const float* __restrict__ gb1,
                      const float* __restrict__ gw2, const float* __restrict__ gb2,
                      const float* __restrict__ lw,  const float* __restrict__ lb,
                      float* __restrict__ out)
{
    __shared__ __align__(16) _Float16 sX[ROWS * XSTR];    // 20KB: x -> H1 -> H2
    __shared__ __align__(16) _Float16 sB[SROWS * Y1STR];  // 27KB: y1 -> Z -> AHAT@H1
    __shared__ __align__(16) _Float16 sC[ROWS * ZSTR];    //  9KB: y2
    __shared__ float sAH[64];

    const int tid    = threadIdx.x;
    const int wave   = tid >> 5;
    const int lane   = tid & 31;
    const int lanehi = lane >> 4;
    const int lanelo = lane & 15;
    const int bbase  = blockIdx.x * TBATCH;

    // ---- AHAT = D^-1/2 (A+I) D^-1/2 for the fixed 8-node graph ----
    if (tid < 64) {
        const unsigned char adj[8] = {0x03, 0x47, 0x2E, 0x1C, 0x38, 0x74, 0xE2, 0xC0};
        const float rd[8] = {0.70710678f, 0.5f, 0.5f, 0.57735027f,
                             0.57735027f, 0.5f, 0.5f, 0.70710678f};
        int i = tid >> 3, j = tid & 7;
        sAH[tid] = ((adj[i] >> j) & 1) ? rd[i] * rd[j] : 0.0f;
    }

    // ---- stage data tile [8,5,8,32] f32 -> LDS f16 x[row][t][obs] (float2) ----
    {
        const float2* d2 = (const float2*)(data + (size_t)bbase * 1280);
        #pragma unroll
        for (int k = 0; k < 20; ++k) {                // 5120 float2 / 256 threads
            int idx2 = tid + (k << 8);
            int bl   = idx2 / 640;
            int rem  = idx2 - bl * 640;
            int l    = rem >> 7;
            int rem2 = rem & 127;
            int node = rem2 >> 4;
            int obs  = (rem2 & 15) << 1;
            float2 v = d2[idx2];
            *(unsigned int*)&sX[(bl * N_NODES + node) * XSTR + l * 32 + obs] =
                f2h2(v.x, v.y);
        }
    }
    __syncthreads();

    // ========== Conv1: [192 x 96] @ [96 x 64], K' = tap*32 + in_channel =======
    {
        int ntile = wave & 3;
        int c     = ntile * 16 + lanelo;              // output channel
        float bias = b1[c];
        v16h bf[3];
        #pragma unroll
        for (int s = 0; s < 3; ++s)                   // s = tap kk
            #pragma unroll
            for (int j = 0; j < 8; ++j)
                #pragma unroll
                for (int h = 0; h < 2; ++h) {
                    int i = KFB(j, h);                // input channel
                    bf[s][2 * j + h] = (_Float16)w1[c * 96 + i * 3 + s];
                }
        int mstart = (wave >> 2) * 6;
        for (int mt = mstart; mt < mstart + 6; ++mt) {
            int mp = mt * 16 + lanelo;                // super-row (r, t)
            int r  = mp / 3, t = mp - 3 * r;
            v8f acc = {0.f, 0.f, 0.f, 0.f, 0.f, 0.f, 0.f, 0.f};
            #pragma unroll
            for (int s = 0; s < 3; ++s) {
                v16h af = LOAD_AFRAG(&sX[r * XSTR + (t + s) * 32]);
                acc = __builtin_amdgcn_wmma_f32_16x16x32_f16(
                          false, af, false, bf[s], (short)0, acc, false, false);
            }
            #pragma unroll
            for (int j = 0; j < 8; ++j) {
                float v = fmaxf(acc[j] + bias, 0.0f);
                sB[(mt * 16 + j + (lanehi << 3)) * Y1STR + c] = (_Float16)v;
            }
        }
    }
    __syncthreads();

    // ========== Conv2: [64 x 192] @ [192 x 64], K' = tap*64 + in_channel ======
    {
        int ntile = wave & 3;
        int c     = ntile * 16 + lanelo;
        float bias = b2[c];
        v16h bf[6];
        #pragma unroll
        for (int s = 0; s < 6; ++s) {                 // tap = s>>1, half = s&1
            int t = s >> 1;
            #pragma unroll
            for (int j = 0; j < 8; ++j)
                #pragma unroll
                for (int h = 0; h < 2; ++h) {
                    int i = ((s & 1) << 5) + KFB(j, h);
                    bf[s][2 * j + h] = (_Float16)w2[c * 192 + i * 3 + t];
                }
        }
        int mstart = (wave >> 2) * 2;
        for (int mt = mstart; mt < mstart + 2; ++mt) {
            int row = mt * 16 + lanelo;
            v8f acc = {0.f, 0.f, 0.f, 0.f, 0.f, 0.f, 0.f, 0.f};
            #pragma unroll
            for (int s = 0; s < 6; ++s) {
                int t = s >> 1;
                v16h af = LOAD_AFRAG(&sB[(row * 3 + t) * Y1STR + ((s & 1) << 5)]);
                acc = __builtin_amdgcn_wmma_f32_16x16x32_f16(
                          false, af, false, bf[s], (short)0, acc, false, false);
            }
            #pragma unroll
            for (int j = 0; j < 8; ++j) {
                float v = fmaxf(acc[j] + bias, 0.0f);
                sC[(mt * 16 + j + (lanehi << 3)) * ZSTR + c] = (_Float16)v;
            }
        }
    }
    __syncthreads();

    // ========== Z = AHAT @ y2 (8x8 node mixing, paired channels) ==============
    for (int e = tid; e < ROWS * 32; e += 256) {
        int row = e >> 5, cp = (e & 31) << 1;
        int b = row >> 3, n = row & 7;
        float s0 = 0.f, s1 = 0.f;
        #pragma unroll
        for (int m = 0; m < 8; ++m) {
            float2 v = h2f2(*(const unsigned int*)&sC[(b * 8 + m) * ZSTR + cp]);
            float a = sAH[n * 8 + m];
            s0 += a * v.x; s1 += a * v.y;
        }
        *(unsigned int*)&sB[row * ZSTR + cp] = f2h2(s0, s1);
    }
    __syncthreads();

    // ========== H1 = relu(Z @ gw1 + gb1): [64 x 64] @ [64 x 128] ==============
    {
        int c = wave * 16 + lanelo;
        float bias = gb1[c];
        v16h bf[2];
        #pragma unroll
        for (int s = 0; s < 2; ++s)
            #pragma unroll
            for (int j = 0; j < 8; ++j)
                #pragma unroll
                for (int h = 0; h < 2; ++h)
                    bf[s][2 * j + h] = (_Float16)gw1[(s * 32 + KFB(j, h)) * 128 + c];
        for (int mt = 0; mt < 4; ++mt) {
            int row = mt * 16 + lanelo;
            v8f acc = {0.f, 0.f, 0.f, 0.f, 0.f, 0.f, 0.f, 0.f};
            #pragma unroll
            for (int s = 0; s < 2; ++s) {
                v16h af = LOAD_AFRAG(&sB[row * ZSTR + s * 32]);
                acc = __builtin_amdgcn_wmma_f32_16x16x32_f16(
                          false, af, false, bf[s], (short)0, acc, false, false);
            }
            #pragma unroll
            for (int j = 0; j < 8; ++j) {
                float v = fmaxf(acc[j] + bias, 0.0f);
                sX[(mt * 16 + j + (lanehi << 3)) * HSTR + c] = (_Float16)v;
            }
        }
    }
    __syncthreads();

    // ========== AZ = AHAT @ H1 (paired channels) ==============================
    for (int e = tid; e < ROWS * 64; e += 256) {
        int row = e >> 6, cp = (e & 63) << 1;
        int b = row >> 3, n = row & 7;
        float s0 = 0.f, s1 = 0.f;
        #pragma unroll
        for (int m = 0; m < 8; ++m) {
            float2 v = h2f2(*(const unsigned int*)&sX[(b * 8 + m) * HSTR + cp]);
            float a = sAH[n * 8 + m];
            s0 += a * v.x; s1 += a * v.y;
        }
        *(unsigned int*)&sB[row * HSTR + cp] = f2h2(s0, s1);
    }
    __syncthreads();

    // ========== H2 = relu(AZ @ gw2 + gb2): [64 x 128] @ [128 x 128] ===========
    {
        int c = wave * 16 + lanelo;
        float bias = gb2[c];
        v16h bf[4];
        #pragma unroll
        for (int s = 0; s < 4; ++s)
            #pragma unroll
            for (int j = 0; j < 8; ++j)
                #pragma unroll
                for (int h = 0; h < 2; ++h)
                    bf[s][2 * j + h] = (_Float16)gw2[(s * 32 + KFB(j, h)) * 128 + c];
        for (int mt = 0; mt < 4; ++mt) {
            int row = mt * 16 + lanelo;
            v8f acc = {0.f, 0.f, 0.f, 0.f, 0.f, 0.f, 0.f, 0.f};
            #pragma unroll
            for (int s = 0; s < 4; ++s) {
                v16h af = LOAD_AFRAG(&sB[row * HSTR + s * 32]);
                acc = __builtin_amdgcn_wmma_f32_16x16x32_f16(
                          false, af, false, bf[s], (short)0, acc, false, false);
            }
            #pragma unroll
            for (int j = 0; j < 8; ++j) {
                float v = fmaxf(acc[j] + bias, 0.0f);
                sX[(mt * 16 + j + (lanehi << 3)) * HSTR + c] = (_Float16)v;
            }
        }
    }
    __syncthreads();

    // ===== head: tanh(H2 @ lw + lb), mean over 8 nodes, write [b,8] ===========
    if (wave < 4) {
        int mt = wave;                                // 16 rows = 2 batch elements
        int n  = lanelo;
        float bias = (n < 8) ? lb[n] : 0.0f;
        v16h bf[4];
        #pragma unroll
        for (int s = 0; s < 4; ++s)
            #pragma unroll
            for (int j = 0; j < 8; ++j)
                #pragma unroll
                for (int h = 0; h < 2; ++h) {
                    int Kg = s * 32 + KFB(j, h);
                    bf[s][2 * j + h] = (n < 8) ? (_Float16)lw[Kg * 8 + n]
                                               : (_Float16)0.0f;
                }
        int row = mt * 16 + lanelo;
        v8f acc = {0.f, 0.f, 0.f, 0.f, 0.f, 0.f, 0.f, 0.f};
        #pragma unroll
        for (int s = 0; s < 4; ++s) {
            v16h af = LOAD_AFRAG(&sX[row * HSTR + s * 32]);
            acc = __builtin_amdgcn_wmma_f32_16x16x32_f16(
                      false, af, false, bf[s], (short)0, acc, false, false);
        }
        float ssum = 0.0f;
        #pragma unroll
        for (int j = 0; j < 8; ++j)
            ssum += tanhf(acc[j] + bias);
        int b = bbase + mt * 2 + lanehi;              // rows j = nodes of batch b
        if (n < 8) out[b * 8 + n] = ssum * 0.125f;
    }
}

extern "C" void kernel_launch(void* const* d_in, const int* in_sizes, int n_in,
                              void* d_out, int out_size, void* d_ws, size_t ws_size,
                              hipStream_t stream) {
    const float* data = (const float*)d_in[0];
    const float* w1   = (const float*)d_in[1];
    const float* b1   = (const float*)d_in[2];
    const float* w2   = (const float*)d_in[3];
    const float* b2   = (const float*)d_in[4];
    const float* gw1  = (const float*)d_in[5];
    const float* gb1  = (const float*)d_in[6];
    const float* gw2  = (const float*)d_in[7];
    const float* gb2  = (const float*)d_in[8];
    const float* lw   = (const float*)d_in[9];
    const float* lb   = (const float*)d_in[10];

    int B    = in_sizes[0] / (LSEQ * N_NODES * OBS);   // 32768
    int grid = B / TBATCH;                             // 4096 workgroups
    gcn_policy_fused<<<grid, 256, 0, stream>>>(
        data, w1, b1, w2, b2, gw1, gb1, gw2, gb2, lw, lb, (float*)d_out);
}
